// GIN_NC_37752762532359
// MI455X (gfx1250) — compile-verified, matrix operand order checked by reading, not run on hardware
//
#include <hip/hip_runtime.h>

typedef __attribute__((ext_vector_type(16))) __bf16 v16bf;
typedef __attribute__((ext_vector_type(8)))  __bf16 v8bf;
typedef __attribute__((ext_vector_type(8)))  float  v8f;

#define HDIM 64

// ---------------------------------------------------------------------------
// z[i] = (1 + eps[epsIdx]) * h[i]   (initializes the aggregation buffer)
// ---------------------------------------------------------------------------
__global__ void init_scale(const float* __restrict__ h,
                           const float* __restrict__ eps, int epsIdx,
                           float* __restrict__ agg, long long total) {
  long long i = (long long)blockIdx.x * blockDim.x + threadIdx.x;
  if (i >= total) return;
  float s = 1.0f + eps[epsIdx];
  agg[i] = s * h[i];
}

// ---------------------------------------------------------------------------
// Edge-parallel scatter-add: agg[dst[e]][:] += h[src[e]][:]
// One thread per (edge, 4 features); gathers + atomics stay L2-resident.
// ---------------------------------------------------------------------------
__global__ void edge_agg(const float* __restrict__ h,
                         const int* __restrict__ src,
                         const int* __restrict__ dst,
                         float* __restrict__ agg,
                         long long nEdges, int F) {
  int chunks = F >> 2;
  long long tid = (long long)blockIdx.x * blockDim.x + threadIdx.x;
  long long e = tid / chunks;
  if (e >= nEdges) return;
  int c = (int)(tid - e * (long long)chunks);
  int s = src[e];
  int d = dst[e];
  const float4 v = *(const float4*)(h + (long long)s * F + c * 4);
  float* p = agg + (long long)d * F + c * 4;
  atomicAdd(p + 0, v.x);
  atomicAdd(p + 1, v.y);
  atomicAdd(p + 2, v.z);
  atomicAdd(p + 3, v.w);
}

// ---------------------------------------------------------------------------
// Repack a weight matrix W [K][C] (fp32, row-major) into the exact wave32
// WMMA B-fragment layout as bf16:  out[((kt*4 + wc)*32 + lane)*16 + i]
// with k = kt*32 + (i&7) + (lane>=16)*8 + (i>=8)*16, col = wc*16 + (lane&15).
// Columns >= C (head: C=40) are zero-padded.
// ---------------------------------------------------------------------------
__global__ void pack_B(const float* __restrict__ W, __bf16* __restrict__ out,
                       int K, int C) {
  int tid = blockIdx.x * blockDim.x + threadIdx.x;
  int total = K * HDIM;               // (K/32)*4*32*16
  if (tid >= total) return;
  int i    = tid & 15;
  int lane = (tid >> 4) & 31;
  int wc   = (tid >> 9) & 3;
  int kt   = tid >> 11;
  int hi   = lane >> 4;
  int l15  = lane & 15;
  int k   = kt * 32 + (i & 7) + hi * 8 + ((i >= 8) ? 16 : 0);
  int col = wc * 16 + l15;
  float v = (col < C) ? W[(long long)k * C + col] : 0.0f;
  out[tid] = (__bf16)v;
}

// ---------------------------------------------------------------------------
// Fused 2-layer MLP via v_wmma_f32_16x16x32_bf16.
//   t1  = relu(Z @ W1 + B1)            (Z: [N,K1] fp32; Wb1: packed bf16)
//   out = relu(t1 @ W2 + B2) -> BN     (useBN, C2=64)
//   out = t1 @ W2 + B2                 (head, C2=40; Wb2 zero-padded)
// Block: 256 threads = 8 waves; wave (wr,wc) owns a 16x16 tile.
// A fragments: 4x global_load_b128 (fp32) + pk converts.
// B fragments: one 32-byte vector load from the packed buffer.
// Intermediate tile staged in LDS as bf16; GEMM2 A via 2x ds_load_b128.
// ---------------------------------------------------------------------------
__global__ __launch_bounds__(256) void mlp_wmma(
    const float* __restrict__ Z, int K1,
    const __bf16* __restrict__ Wb1, const float* __restrict__ B1,
    const __bf16* __restrict__ Wb2, const float* __restrict__ B2,
    const float* __restrict__ gamma, const float* __restrict__ beta,
    const float* __restrict__ mean, const float* __restrict__ var,
    float* __restrict__ out, int N, int C2, int useBN) {
  __shared__ alignas(16) __bf16 lds_t1[32][HDIM + 8];   // row stride 144B

  const int lane = threadIdx.x & 31;
  const int wave = threadIdx.x >> 5;
  const int wr   = wave >> 2;      // row tile 0..1
  const int wc   = wave & 3;       // col tile 0..3
  const int hi   = (lane >> 4) & 1;
  const int l15  = lane & 15;
  const int rowBase = blockIdx.x * 32;

  // ---- GEMM1 ----
  int aRow = rowBase + wr * 16 + l15;
  if (aRow >= N) aRow = N - 1;     // N % 32 == 0 here; safety clamp
  const float* arow = Z + (long long)aRow * K1 + hi * 8;

  v8f acc = {};
  for (int kt = 0; kt < (K1 >> 5); ++kt) {
    const int kk = kt * 32;
    float4 p0 = *(const float4*)(arow + kk + 0);
    float4 p1 = *(const float4*)(arow + kk + 4);
    float4 p2 = *(const float4*)(arow + kk + 16);
    float4 p3 = *(const float4*)(arow + kk + 20);
    v16bf a;
    a[0]  = (__bf16)p0.x; a[1]  = (__bf16)p0.y; a[2]  = (__bf16)p0.z; a[3]  = (__bf16)p0.w;
    a[4]  = (__bf16)p1.x; a[5]  = (__bf16)p1.y; a[6]  = (__bf16)p1.z; a[7]  = (__bf16)p1.w;
    a[8]  = (__bf16)p2.x; a[9]  = (__bf16)p2.y; a[10] = (__bf16)p2.z; a[11] = (__bf16)p2.w;
    a[12] = (__bf16)p3.x; a[13] = (__bf16)p3.y; a[14] = (__bf16)p3.z; a[15] = (__bf16)p3.w;
    v16bf b = *(const v16bf*)(Wb1 + (((long long)kt * 4 + wc) * 32 + lane) * 16);
    acc = __builtin_amdgcn_wmma_f32_16x16x32_bf16(
        false, a, false, b, (short)0, acc, false, false);
  }

  // epilogue 1: bias + relu -> bf16 tile in LDS
  {
    int col = wc * 16 + l15;
    float bias = B1[col];
#pragma unroll
    for (int m = 0; m < 8; ++m) {
      int row = wr * 16 + hi * 8 + m;
      float t = acc[m] + bias;
      t = t > 0.0f ? t : 0.0f;
      lds_t1[row][col] = (__bf16)t;
    }
  }
  __syncthreads();

  // ---- GEMM2 ----
  v8f acc2 = {};
#pragma unroll
  for (int kt = 0; kt < 2; ++kt) {
    const int kk = kt * 32;
    v8bf q0 = *(const v8bf*)&lds_t1[wr * 16 + l15][kk + hi * 8];
    v8bf q1 = *(const v8bf*)&lds_t1[wr * 16 + l15][kk + 16 + hi * 8];
    v16bf a;
#pragma unroll
    for (int i = 0; i < 8; ++i) { a[i] = q0[i]; a[8 + i] = q1[i]; }
    v16bf b = *(const v16bf*)(Wb2 + (((long long)kt * 4 + wc) * 32 + lane) * 16);
    acc2 = __builtin_amdgcn_wmma_f32_16x16x32_bf16(
        false, a, false, b, (short)0, acc2, false, false);
  }

  // epilogue 2: bias (+ relu + eval-mode BN for GIN layers)
  {
    int col = wc * 16 + l15;
    if (col < C2) {
      float bias = B2[col];
      float g = 1.0f, bt = 0.0f, mu = 0.0f, rs = 1.0f;
      if (useBN) {
        g  = gamma[col];
        bt = beta[col];
        mu = mean[col];
        rs = rsqrtf(var[col] + 1e-5f);
      }
#pragma unroll
      for (int m = 0; m < 8; ++m) {
        int row = rowBase + wr * 16 + hi * 8 + m;
        if (row < N) {
          float t = acc2[m] + bias;
          if (useBN) {
            t = t > 0.0f ? t : 0.0f;
            t = g * (t - mu) * rs + bt;
          }
          out[(long long)row * C2 + col] = t;
        }
      }
    }
  }
}

// ---------------------------------------------------------------------------
extern "C" void kernel_launch(void* const* d_in, const int* in_sizes, int n_in,
                              void* d_out, int out_size, void* d_ws, size_t ws_size,
                              hipStream_t stream) {
  const float* x        = (const float*)d_in[0];   // [N,128]
  const int*   edge     = (const int*)d_in[1];     // [2,E]
  const float* eps      = (const float*)d_in[2];   // [3]
  const float* c1_w1    = (const float*)d_in[3];   // [128,64]
  const float* c1_b1    = (const float*)d_in[4];
  const float* c1_w2    = (const float*)d_in[5];   // [64,64]
  const float* c1_b2    = (const float*)d_in[6];
  const float* c1_gamma = (const float*)d_in[7];
  const float* c1_beta  = (const float*)d_in[8];
  const float* c1_mean  = (const float*)d_in[9];
  const float* c1_var   = (const float*)d_in[10];
  const float* cw1      = (const float*)d_in[11];  // [2,64,64]
  const float* cb1      = (const float*)d_in[12];  // [2,64]
  const float* cw2      = (const float*)d_in[13];
  const float* cb2      = (const float*)d_in[14];
  const float* cgamma   = (const float*)d_in[15];
  const float* cbeta    = (const float*)d_in[16];
  const float* cmean    = (const float*)d_in[17];
  const float* cvar     = (const float*)d_in[18];
  const float* lin1_w   = (const float*)d_in[19];  // [64,64]
  const float* lin1_b   = (const float*)d_in[20];
  const float* lin2_w   = (const float*)d_in[21];  // [64,40]
  const float* lin2_b   = (const float*)d_in[22];

  const long long N = in_sizes[0] / 128;           // 100000
  const long long E = in_sizes[1] / 2;             // 1600000
  const int NC = 40;

  const int* src = edge;
  const int* dst = edge + E;

  // workspace: zA [N,128] f32 | hB [N,64] f32 | hC [N,64] f32 | packed bf16 B
  float* zA = (float*)d_ws;
  float* hB = zA + N * 128;
  float* hC = hB + N * HDIM;
  __bf16* pk = (__bf16*)(hC + N * HDIM);
  __bf16* pb_c1w1 = pk;                  // 128*64
  __bf16* pb_c1w2 = pb_c1w1 + 128 * 64;  // 64*64
  __bf16* pb_w1a  = pb_c1w2 + 64 * 64;
  __bf16* pb_w2a  = pb_w1a  + 64 * 64;
  __bf16* pb_w1b  = pb_w2a  + 64 * 64;
  __bf16* pb_w2b  = pb_w1b  + 64 * 64;
  __bf16* pb_l1   = pb_w2b  + 64 * 64;
  __bf16* pb_l2   = pb_l1   + 64 * 64;   // 64x40 zero-padded to 64x64

  const int TB = 256;
  const int mlpGrid = (int)((N + 31) / 32);
  auto blocksFor = [](long long t, int tb) { return (int)((t + tb - 1) / tb); };

  // ---- one-time weight repack into WMMA B-fragment layout (bf16) ----
  pack_B<<<blocksFor(128 * 64, TB), TB, 0, stream>>>(c1_w1, pb_c1w1, 128, 64);
  pack_B<<<blocksFor(64 * 64, TB), TB, 0, stream>>>(c1_w2, pb_c1w2, 64, 64);
  pack_B<<<blocksFor(64 * 64, TB), TB, 0, stream>>>(cw1, pb_w1a, 64, 64);
  pack_B<<<blocksFor(64 * 64, TB), TB, 0, stream>>>(cw2, pb_w2a, 64, 64);
  pack_B<<<blocksFor(64 * 64, TB), TB, 0, stream>>>(cw1 + 64 * 64, pb_w1b, 64, 64);
  pack_B<<<blocksFor(64 * 64, TB), TB, 0, stream>>>(cw2 + 64 * 64, pb_w2b, 64, 64);
  pack_B<<<blocksFor(64 * 64, TB), TB, 0, stream>>>(lin1_w, pb_l1, 64, 64);
  pack_B<<<blocksFor(64 * 64, TB), TB, 0, stream>>>(lin2_w, pb_l2, 64, NC);

  // ---- GIN layer 1 (128 -> 64) ----
  init_scale<<<blocksFor(N * 128, TB), TB, 0, stream>>>(x, eps, 0, zA, N * 128);
  edge_agg<<<blocksFor(E * 32, TB), TB, 0, stream>>>(x, src, dst, zA, E, 128);
  mlp_wmma<<<mlpGrid, TB, 0, stream>>>(zA, 128, pb_c1w1, c1_b1, pb_c1w2, c1_b2,
                                       c1_gamma, c1_beta, c1_mean, c1_var,
                                       hB, (int)N, HDIM, 1);

  // ---- GIN layer 2 (64 -> 64) ----
  init_scale<<<blocksFor(N * HDIM, TB), TB, 0, stream>>>(hB, eps, 1, zA, N * HDIM);
  edge_agg<<<blocksFor(E * 16, TB), TB, 0, stream>>>(hB, src, dst, zA, E, HDIM);
  mlp_wmma<<<mlpGrid, TB, 0, stream>>>(zA, HDIM, pb_w1a, cb1, pb_w2a, cb2,
                                       cgamma, cbeta, cmean, cvar,
                                       hC, (int)N, HDIM, 1);

  // ---- GIN layer 3 (64 -> 64) ----
  init_scale<<<blocksFor(N * HDIM, TB), TB, 0, stream>>>(hC, eps, 2, zA, N * HDIM);
  edge_agg<<<blocksFor(E * 16, TB), TB, 0, stream>>>(hC, src, dst, zA, E, HDIM);
  mlp_wmma<<<mlpGrid, TB, 0, stream>>>(zA, HDIM, pb_w1b, cb1 + HDIM, pb_w2b, cb2 + HDIM,
                                       cgamma + HDIM, cbeta + HDIM,
                                       cmean + HDIM, cvar + HDIM,
                                       hB, (int)N, HDIM, 1);

  // ---- Head: relu(h@lin1+b1) @ lin2 + b2 -> [N,40] ----
  mlp_wmma<<<mlpGrid, TB, 0, stream>>>(hB, HDIM, pb_l1, lin1_b, pb_l2, lin2_b,
                                       nullptr, nullptr, nullptr, nullptr,
                                       (float*)d_out, (int)N, NC, 0);
}
